// Model14v2_9620726743228
// MI455X (gfx1250) — compile-verified, matrix operand order checked by reading, not run on hardware
//
#include <hip/hip_runtime.h>
#include <cstddef>

// ---------------- WMMA types (CDNA5, wave32) ----------------
typedef __attribute__((ext_vector_type(16))) _Float16 v16h;
typedef __attribute__((ext_vector_type(8)))  _Float16 v8h;
typedef __attribute__((ext_vector_type(2)))  _Float16 v2h;
typedef __attribute__((ext_vector_type(8)))  float    v8f;

// ---------------- A-operand fetch functors ----------------
struct FPlain {
  const float* A; int ld;
  __device__ float operator()(int r, int k) const { return A[(size_t)r * ld + k]; }
};
// [graph(15) | xcat(up to 150)] row-concat
struct FGX {
  const float* g; int cg; const float* xc;
  __device__ float operator()(int r, int k) const {
    return (k < cg) ? g[(size_t)r * cg + k] : xc[(size_t)r * 150 + (k - cg)];
  }
};
// [data(c0) | xcat[i1[r]](150) | xcat[i2[r]](150)]
struct FGatherPair {
  const float* d; int ldd, c0; const float* xc; const int* i1; const int* i2;
  __device__ float operator()(int r, int k) const {
    if (k < c0) return d[(size_t)r * ldd + k];
    k -= c0;
    if (k < 150) return xc[(size_t)i1[r] * 150 + k];
    return xc[(size_t)i2[r] * 150 + (k - 150)];
  }
};
// [data(c0) | xcat[i1[r]](150)]
struct FGatherOne {
  const float* d; int ldd, c0; const float* xc; const int* i1;
  __device__ float operator()(int r, int k) const {
    return (k < c0) ? d[(size_t)r * ldd + k] : xc[(size_t)i1[r] * 150 + (k - c0)];
  }
};

// ---------------- WMMA GEMM workhorse ----------------
// out[M x Ncols] = act(A[M x Kdim] @ W[Kdim x Ncols] + bias)
// 128 threads = 4 waves. Block covers 64 rows x (16*NCT) cols:
//   wave w owns rows [row0+16w, +16); NCT col tiles fully unrolled.
// Staging is branch-free: indices clamped so loads are always valid and
// batchable; OOB lanes zeroed by select afterwards (no exec-mask diamonds,
// one wait for the whole burst instead of per-element waits).
// A and W^T staged in LDS as f16, 40-half (80B) pitch: rows 16B-aligned,
// start bank = r*20 mod 64 distinct over 16 rows -> conflict-free ds_load_b128.
// act: 0=none, 1=relu, 2=leaky(0.01)
template <class AF, int NCT>
__global__ __launch_bounds__(128) void gemm_wmma_kernel(
    AF af, const float* __restrict__ W, int ldw, const float* __restrict__ bias,
    int M, int Kdim, int Ncols, float* __restrict__ out, int ldo, int act) {
  __shared__ __align__(16) _Float16 aS[64][40];
  __shared__ __align__(16) _Float16 wT[64][40];
  const int tid  = threadIdx.x;
  const int wave = tid >> 5;
  const int lane = tid & 31;
  const int half = lane >> 4;
  const int l16  = lane & 15;
  const int row0 = blockIdx.x * 64;
  const int nb64 = blockIdx.y * 64;

  v8f acc[NCT];
#pragma unroll
  for (int ct = 0; ct < NCT; ++ct) acc[ct] = {};

  for (int k0 = 0; k0 < Kdim; k0 += 32) {
    // stage A tile 64x32 -> f16 LDS; clamped indices, unconditional loads
#pragma unroll
    for (int it = 0; it < 8; ++it) {
      int i = tid + it * 128;
      int m = i >> 4, kp = (i & 15) << 1;
      int r = row0 + m, k = k0 + kp;
      int rr  = r < M ? r : (M - 1);
      int kc0 = k < Kdim ? k : (Kdim - 1);
      int kc1 = (k + 1) < Kdim ? (k + 1) : (Kdim - 1);
      float v0 = af(rr, kc0);
      float v1 = af(rr, kc1);
      if (!(r < M && k < Kdim)) v0 = 0.f;
      if (!(r < M && (k + 1) < Kdim)) v1 = 0.f;
      v2h h; h[0] = (_Float16)v0; h[1] = (_Float16)v1;
      *(v2h*)&aS[m][kp] = h;
    }
    // stage W tile 32x64 -> transposed f16 LDS (coalesced reads along n)
#pragma unroll
    for (int it = 0; it < 16; ++it) {
      int i = tid + it * 128;
      int kk = i >> 6, n = i & 63;
      int k = k0 + kk, c = nb64 + n;
      int kc = k < Kdim ? k : (Kdim - 1);
      int cc = c < Ncols ? c : (Ncols - 1);
      float v = W[(size_t)kc * ldw + cc];
      if (!(k < Kdim && c < Ncols)) v = 0.f;
      wT[n][kk] = (_Float16)v;
    }
    // prefetch next K-chunk of W (global_prefetch_b8)
    if (k0 + 32 < Kdim && tid < 32 && (k0 + 32 + tid) < Kdim)
      __builtin_prefetch(&W[(size_t)(k0 + 32 + tid) * ldw + nb64], 0, 1);
    __syncthreads();
    // 16-bit A/B 16x32 fragment: element e -> K = e%8 + 8*half + 16*(e/8)
    // -> two contiguous 16B LDS loads at half-offsets {8*half, 16+8*half}
    v8h alo = *(const v8h*)&aS[wave * 16 + l16][8 * half];
    v8h ahi = *(const v8h*)&aS[wave * 16 + l16][16 + 8 * half];
    v16h afr = __builtin_shufflevector(alo, ahi, 0, 1, 2, 3, 4, 5, 6, 7,
                                       8, 9, 10, 11, 12, 13, 14, 15);
#pragma unroll
    for (int ct = 0; ct < NCT; ++ct) {
      v8h blo = *(const v8h*)&wT[ct * 16 + l16][8 * half];
      v8h bhi = *(const v8h*)&wT[ct * 16 + l16][16 + 8 * half];
      v16h bfr = __builtin_shufflevector(blo, bhi, 0, 1, 2, 3, 4, 5, 6, 7,
                                         8, 9, 10, 11, 12, 13, 14, 15);
      acc[ct] = __builtin_amdgcn_wmma_f32_16x16x32_f16(
          false, afr, false, bfr, (short)0, acc[ct], false, false);
    }
    __syncthreads();
  }
  // C/D layout: lane = half*16 + n(0..15); VGPR r -> M = r + 8*half
#pragma unroll
  for (int ct = 0; ct < NCT; ++ct) {
    int col = nb64 + ct * 16 + l16;
    if (col < Ncols) {
      float b = bias ? bias[col] : 0.f;
#pragma unroll
      for (int r = 0; r < 8; ++r) {
        int m = row0 + wave * 16 + r + 8 * half;
        if (m < M) {
          float v = acc[ct][r] + b;
          if (act == 1) v = fmaxf(v, 0.f);
          else if (act == 2) v = (v > 0.f) ? v : 0.01f * v;
          out[(size_t)m * ldo + col] = v;
        }
      }
    }
  }
}

template <class AF>
static void launch_gemm(AF af, const float* W, int ldw, const float* b, int M,
                        int K, int Nc, float* out, int ldo, int act,
                        hipStream_t s) {
  dim3 grid((M + 63) / 64, (Nc + 63) / 64);
  int nct = ((Nc < 64 ? Nc : 64) + 15) / 16;  // uniform per launch
  switch (nct) {
    case 1: gemm_wmma_kernel<AF, 1><<<grid, 128, 0, s>>>(af, W, ldw, b, M, K, Nc, out, ldo, act); break;
    case 2: gemm_wmma_kernel<AF, 2><<<grid, 128, 0, s>>>(af, W, ldw, b, M, K, Nc, out, ldo, act); break;
    case 3: gemm_wmma_kernel<AF, 3><<<grid, 128, 0, s>>>(af, W, ldw, b, M, K, Nc, out, ldo, act); break;
    default: gemm_wmma_kernel<AF, 4><<<grid, 128, 0, s>>>(af, W, ldw, b, M, K, Nc, out, ldo, act); break;
  }
}

// ---------------- weight packing (fuse q/k/v/skip into one GEMM) ----------------
__global__ void pack_w_kernel(float* __restrict__ dst, int ldd, int coff,
                              const float* __restrict__ src, int K, int C) {
  int i = blockIdx.x * blockDim.x + threadIdx.x;
  if (i >= K * C) return;
  int k = i / C, c = i % C;
  dst[(size_t)k * ldd + coff + c] = src[i];
}
__global__ void pack_b_kernel(float* __restrict__ dst, int coff,
                              const float* __restrict__ src, int C) {
  int i = blockIdx.x * blockDim.x + threadIdx.x;
  if (i < C) dst[coff + i] = src[i];
}

// ---------------- fills ----------------
__global__ void fill_kernel(float* p, float v, size_t n) {
  size_t i = (size_t)blockIdx.x * blockDim.x + threadIdx.x;
  if (i < n) p[i] = v;
}
static void fill(float* p, float v, size_t n, hipStream_t s) {
  fill_kernel<<<(unsigned)((n + 255) / 256), 256, 0, s>>>(p, v, n);
}

// ---------------- edge attention ----------------
__device__ inline void atomicMaxFloat(float* addr, float v) {
  if (v >= 0.f) atomicMax((int*)addr, __float_as_int(v));
  else          atomicMin((unsigned int*)addr, __float_as_uint(v));
}

__global__ void edge_alpha_kernel(const int* __restrict__ src, const int* __restrict__ dst,
                                  const float* __restrict__ q, const float* __restrict__ k, int ld,
                                  float* __restrict__ alpha, float* __restrict__ nmax, int E) {
  int e = blockIdx.x * blockDim.x + threadIdx.x;
  if (e >= E) return;
  int s = src[e], d = dst[e];
  const float* qa = q + (size_t)d * ld;
  const float* ka = k + (size_t)s * ld;
  float acc = 0.f;
#pragma unroll 10
  for (int i = 0; i < 50; ++i) acc += qa[i] * ka[i];
  acc *= 0.14142135623730950f;  // 1/sqrt(50)
  alpha[e] = acc;
  atomicMaxFloat(&nmax[d], acc);
}

__global__ void edge_expsum_kernel(const int* __restrict__ dst, const float* __restrict__ alpha,
                                   const float* __restrict__ nmax, float* __restrict__ eexp,
                                   float* __restrict__ nsum, int E) {
  int e = blockIdx.x * blockDim.x + threadIdx.x;
  if (e >= E) return;
  int d = dst[e];
  float m = nmax[d];
  if (!(m >= -3.0e38f && m <= 3.0e38f)) m = 0.f;  // jnp.where(isfinite(m), m, 0)
  float x = __expf(alpha[e] - m);
  eexp[e] = x;
  atomicAdd(&nsum[d], x);
}

__global__ void edge_scatter_kernel(const int* __restrict__ src, const int* __restrict__ dst,
                                    const float* __restrict__ eexp, const float* __restrict__ nsum,
                                    const float* __restrict__ v, int ldv,
                                    float* __restrict__ outagg, int E) {
  int e = blockIdx.x * blockDim.x + threadIdx.x;
  if (e >= E) return;
  int s = src[e], d = dst[e];
  float a = eexp[e] / nsum[d];
  const float* vr = v + (size_t)s * ldv;
  float* o = outagg + (size_t)d * 50;
#pragma unroll 10
  for (int i = 0; i < 50; ++i) atomicAdd(&o[i], vr[i] * a);
}

// beta-gated skip + ReLU
__global__ void combine_beta_kernel(const float* __restrict__ outagg,
                                    const float* __restrict__ skip, int lds,
                                    const float* __restrict__ Wbeta, float* __restrict__ y, int N) {
  int r = blockIdx.x * blockDim.x + threadIdx.x;
  if (r >= N) return;
  const float* o = outagg + (size_t)r * 50;
  const float* x = skip + (size_t)r * lds;
  float t = 0.f;
#pragma unroll 10
  for (int i = 0; i < 50; ++i)
    t += o[i] * Wbeta[i] + x[i] * Wbeta[50 + i] + (o[i] - x[i]) * Wbeta[100 + i];
  float beta = 1.f / (1.f + __expf(-t));
  float* yo = y + (size_t)r * 50;
#pragma unroll 10
  for (int i = 0; i < 50; ++i) {
    float val = beta * x[i] + (1.f - beta) * o[i];
    yo[i] = fmaxf(val, 0.f);
  }
}

// per-(graph,channel) group norm; batch segments contiguous (rps rows each)
__global__ void gnorm_kernel(const float* __restrict__ x, int ldx, const float* __restrict__ w,
                             const float* __restrict__ b, const float* __restrict__ ms,
                             float* __restrict__ out, int ldo, int rps) {
  int seg = blockIdx.x, c = threadIdx.x;
  if (c >= 50) return;
  const float* xs = x + (size_t)seg * rps * ldx;
  float s = 0.f;
  for (int i = 0; i < rps; ++i) s += xs[(size_t)i * ldx + c];
  float mc = (s / rps) * ms[c];
  float v = 0.f;
  for (int i = 0; i < rps; ++i) { float d = xs[(size_t)i * ldx + c] - mc; v += d * d; }
  v /= rps;
  float inv = rsqrtf(v + 1e-5f);
  float* os = out + (size_t)seg * rps * ldo;
  for (int i = 0; i < rps; ++i)
    os[(size_t)i * ldo + c] = w[c] * (xs[(size_t)i * ldx + c] - mc) * inv + b[c];
}

// global-attention pooling per contiguous segment
__global__ void pool_kernel(const float* __restrict__ gate, int ldg,
                            const float* __restrict__ h, int ldh,
                            float* __restrict__ pooled, int rps) {
  __shared__ float sm[256];
  __shared__ float acc[11];
  int seg = blockIdx.x, tid = threadIdx.x;
  const float* g  = gate + (size_t)seg * rps * ldg;
  const float* hh = h + (size_t)seg * rps * ldh;
  float mx = -3.4e38f;
  for (int i = tid; i < rps; i += 256) mx = fmaxf(mx, g[(size_t)i * ldg]);
  sm[tid] = mx; __syncthreads();
  for (int s = 128; s > 0; s >>= 1) { if (tid < s) sm[tid] = fmaxf(sm[tid], sm[tid + s]); __syncthreads(); }
  mx = sm[0];
  if (tid < 11) acc[tid] = 0.f;
  __syncthreads();
  float lsum = 0.f, lc[10];
#pragma unroll
  for (int c = 0; c < 10; ++c) lc[c] = 0.f;
  for (int i = tid; i < rps; i += 256) {
    float w = __expf(g[(size_t)i * ldg] - mx);
    lsum += w;
#pragma unroll
    for (int c = 0; c < 10; ++c) lc[c] += w * hh[(size_t)i * ldh + c];
  }
  atomicAdd(&acc[10], lsum);
#pragma unroll
  for (int c = 0; c < 10; ++c) atomicAdd(&acc[c], lc[c]);
  __syncthreads();
  if (tid < 10) pooled[(size_t)seg * 10 + tid] = acc[tid] / acc[10];
}

// V = tanh(lin2(relu(lin1([relu(pooled), global]))))
__global__ void value_head_kernel(const float* __restrict__ pooled, const float* __restrict__ gd,
                                  const float* __restrict__ W1, const float* __restrict__ b1,
                                  const float* __restrict__ W2, const float* __restrict__ b2,
                                  float* __restrict__ V, int B) {
  int b = blockIdx.x * blockDim.x + threadIdx.x;
  if (b >= B) return;
  float in[14];
#pragma unroll
  for (int i = 0; i < 10; ++i) in[i] = fmaxf(pooled[(size_t)b * 10 + i], 0.f);
#pragma unroll
  for (int i = 0; i < 4; ++i) in[10 + i] = gd[(size_t)b * 4 + i];
  float s2 = b2[0];
#pragma unroll
  for (int j = 0; j < 10; ++j) {
    float s = b1[j];
#pragma unroll
    for (int i = 0; i < 14; ++i) s += in[i] * W1[i * 10 + j];
    s2 += fmaxf(s, 0.f) * W2[j];
  }
  V[b] = tanhf(s2);
}

// scatter-add 20-wide rows into segment accumulator
__global__ void seg_add20_kernel(const float* __restrict__ rows, const int* __restrict__ seg,
                                 float* __restrict__ tmp, int M) {
  int r = blockIdx.x * blockDim.x + threadIdx.x;
  if (r >= M) return;
  const float* x = rows + (size_t)r * 20;
  float* t = tmp + (size_t)seg[r] * 20;
#pragma unroll
  for (int i = 0; i < 20; ++i) atomicAdd(&t[i], x[i]);
}

// final: relu(tmp) @ Wf + bf, then log_softmax over num_moves per batch row
__global__ void final_logsoftmax_kernel(const float* __restrict__ tmp, const float* __restrict__ Wf,
                                        const float* __restrict__ bf, float* __restrict__ outp, int NM) {
  __shared__ float sh[256];
  int b = blockIdx.x, t = threadIdx.x;
  const float* row = tmp + ((size_t)b * NM + t) * 20;
  float s = bf[0];
#pragma unroll
  for (int i = 0; i < 20; ++i) s += fmaxf(row[i], 0.f) * Wf[i];
  sh[t] = s; __syncthreads();
  for (int st = blockDim.x >> 1; st > 0; st >>= 1) { if (t < st) sh[t] = fmaxf(sh[t], sh[t + st]); __syncthreads(); }
  float mx = sh[0]; __syncthreads();
  sh[t] = __expf(s - mx); __syncthreads();
  for (int st = blockDim.x >> 1; st > 0; st >>= 1) { if (t < st) sh[t] += sh[t + st]; __syncthreads(); }
  float lse = logf(sh[0]);
  outp[(size_t)b * NM + t] = s - mx - lse;
}

// ---------------- driver ----------------
extern "C" void kernel_launch(void* const* d_in, const int* in_sizes, int n_in,
                              void* d_out, int out_size, void* d_ws, size_t ws_size,
                              hipStream_t stream) {
  const float* graph    = (const float*)d_in[0];
  const float* gdata    = (const float*)d_in[1];
  const float* attackd  = (const float*)d_in[2];
  const float* transfd  = (const float*)d_in[3];
  const float* deployd  = (const float*)d_in[4];
  const int*   eidx     = (const int*)d_in[5];
  const int*   asrcs    = (const int*)d_in[7];
  const int*   adsts    = (const int*)d_in[8];
  const int*   tsrcs    = (const int*)d_in[9];
  const int*   tdsts    = (const int*)d_in[10];
  const int*   dtgts    = (const int*)d_in[11];
  const int*   abtch    = (const int*)d_in[12];
  const int*   tbtch    = (const int*)d_in[13];
  const int*   dbtch    = (const int*)d_in[14];
  // params start at 16 (dict-order flattening)
  const int PB = 16;                               // g1 @16, g2 @25, g3 @34
  const int NORM = 43, ATTG = 52, ATTN = 56;
  const int LIN1 = 60, LIN2 = 62, AT1 = 64, AT2 = 66, TR1 = 68, TR2 = 70;
  const int DP1 = 72, DP2 = 74, ACC = 76, FIN = 78;
  auto P = [&](int i) { return (const float*)d_in[i]; };

  const int N  = in_sizes[0] / 15;
  const int B  = in_sizes[1] / 4;
  const int NA = in_sizes[7];
  const int NT = in_sizes[9];
  const int ND = in_sizes[11];
  const int E  = in_sizes[5] / 2;
  const int NM = out_size / B - 1;                 // out = V[B] ++ logp[B*NM]
  const int rps = N / B;
  const int* esrc = eidx;
  const int* edst = eidx + E;

  // workspace layout (floats)
  float* w = (float*)d_ws;
  size_t off = 0;
  auto alloc = [&](size_t n) { float* p = w + off; off += (n + 15) & ~(size_t)15; return p; };
  float* qkvs   = alloc((size_t)N * 200);  // packed [q|k|v|skip] ld=200 (reused: [gate_h|nn_h] ld=100)
  float* outagg = alloc((size_t)N * 50);
  float* ybuf   = alloc((size_t)N * 50);
  float* xcat   = alloc((size_t)N * 150);
  float* alpha  = alloc((size_t)E);
  float* eexp   = alloc((size_t)E);
  float* nmax   = alloc((size_t)N);
  float* nsum   = alloc((size_t)N);
  float* hbuf   = alloc((size_t)N * 10);
  float* gate   = alloc((size_t)N);
  float* pooled = alloc((size_t)B * 10);
  float* wcat   = alloc((size_t)200 * 200);        // packed weights (<=115x200 / 165x100)
  float* bcat   = alloc(256);
  size_t mxh = (size_t)(NA > NT ? (NA > ND ? NA : ND) : (NT > ND ? NT : ND));
  float* hid    = alloc(mxh * 20);
  float* ordbuf = alloc((size_t)(NA + NT + ND) * 20);
  float* orders = alloc((size_t)(NA + NT + ND) * 20);
  float* tmpseg = alloc((size_t)B * NM * 20);
  (void)ws_size; (void)n_in;

  const float NEG_INF = -__builtin_huge_valf();
  const int EB = (E + 255) / 256, NB = (N + 255) / 256;

  auto packw = [&](float* dst, int ldd, int coff, const float* src, int K, int C) {
    pack_w_kernel<<<(K * C + 255) / 256, 256, 0, stream>>>(dst, ldd, coff, src, K, C);
  };
  auto packb = [&](float* dst, int coff, const float* src, int C) {
    pack_b_kernel<<<1, 256, 0, stream>>>(dst, coff, src, C);
  };

  // ---- 3 TransformerConv layers ----
  for (int l = 0; l < 3; ++l) {
    int pb = PB + l * 9;
    int K  = 15 + l * 50;
    // pack [Wq|Wk|Wv|Wskip] -> wcat[K x 200], biases -> bcat[200]
    for (int j = 0; j < 4; ++j) {
      packw(wcat, 200, j * 50, P(pb + 2 * j), K, 50);
      packb(bcat, j * 50, P(pb + 2 * j + 1), 50);
    }
    if (l == 0) launch_gemm(FPlain{graph, 15},    wcat, 200, bcat, N, K, 200, qkvs, 200, 0, stream);
    else        launch_gemm(FGX{graph, 15, xcat}, wcat, 200, bcat, N, K, 200, qkvs, 200, 0, stream);

    fill(nmax, NEG_INF, N, stream);
    fill(nsum, 0.f, N, stream);
    fill(outagg, 0.f, (size_t)N * 50, stream);
    edge_alpha_kernel  <<<EB, 256, 0, stream>>>(esrc, edst, qkvs, qkvs + 50, 200, alpha, nmax, E);
    edge_expsum_kernel <<<EB, 256, 0, stream>>>(edst, alpha, nmax, eexp, nsum, E);
    edge_scatter_kernel<<<EB, 256, 0, stream>>>(esrc, edst, eexp, nsum, qkvs + 100, 200, outagg, E);
    combine_beta_kernel<<<NB, 256, 0, stream>>>(outagg, qkvs + 150, 200, P(pb + 8), ybuf, N);
    gnorm_kernel<<<B, 64, 0, stream>>>(ybuf, 50, P(NORM + l * 3), P(NORM + l * 3 + 1),
                                       P(NORM + l * 3 + 2), xcat + l * 50, 150, rps);
  }

  // ---- global-attention pooling + value head ----
  // pack [W_gate_l1 | W_nn_l1] -> wcat[165 x 100]; one shared-A GEMM (leaky relu)
  packw(wcat, 100, 0,  P(ATTG + 0), 165, 50);
  packw(wcat, 100, 50, P(ATTN + 0), 165, 50);
  packb(bcat, 0,  P(ATTG + 1), 50);
  packb(bcat, 50, P(ATTN + 1), 50);
  launch_gemm(FGX{graph, 15, xcat}, wcat, 100, bcat, N, 165, 100, qkvs, 100, 2, stream);
  launch_gemm(FPlain{qkvs, 100},      P(ATTG + 2), 1,  P(ATTG + 3), N, 50, 1,  gate, 1,  0, stream);
  launch_gemm(FPlain{qkvs + 50, 100}, P(ATTN + 2), 10, P(ATTN + 3), N, 50, 10, hbuf, 10, 0, stream);
  pool_kernel<<<B, 256, 0, stream>>>(gate, 1, hbuf, 10, pooled, rps);
  value_head_kernel<<<(B + 63) / 64, 64, 0, stream>>>(pooled, gdata, P(LIN1), P(LIN1 + 1),
                                                      P(LIN2), P(LIN2 + 1), (float*)d_out, B);

  // ---- order heads (gather fused into WMMA A-fetch) ----
  launch_gemm(FGatherPair{attackd, 52, 52, xcat, asrcs, adsts}, P(AT1), 20, P(AT1 + 1),
              NA, 352, 20, hid, 20, 1, stream);
  launch_gemm(FPlain{hid, 20}, P(AT2), 20, P(AT2 + 1), NA, 20, 20, ordbuf, 20, 0, stream);
  launch_gemm(FGatherPair{transfd, 51, 51, xcat, tsrcs, tdsts}, P(TR1), 20, P(TR1 + 1),
              NT, 351, 20, hid, 20, 1, stream);
  launch_gemm(FPlain{hid, 20}, P(TR2), 20, P(TR2 + 1), NT, 20, 20,
              ordbuf + (size_t)NA * 20, 20, 0, stream);
  launch_gemm(FGatherOne{deployd, 26, 26, xcat, dtgts}, P(DP1), 20, P(DP1 + 1),
              ND, 176, 20, hid, 20, 1, stream);
  launch_gemm(FPlain{hid, 20}, P(DP2), 20, P(DP2 + 1), ND, 20, 20,
              ordbuf + (size_t)(NA + NT) * 20, 20, 0, stream);
  launch_gemm(FPlain{ordbuf, 20}, P(ACC), 20, P(ACC + 1), NA + NT + ND, 20, 20,
              orders, 20, 0, stream);

  fill(tmpseg, 0.f, (size_t)B * NM * 20, stream);
  seg_add20_kernel<<<(NA + 255) / 256, 256, 0, stream>>>(orders, abtch, tmpseg, NA);
  seg_add20_kernel<<<(NT + 255) / 256, 256, 0, stream>>>(orders + (size_t)NA * 20, tbtch, tmpseg, NT);
  seg_add20_kernel<<<(ND + 255) / 256, 256, 0, stream>>>(orders + (size_t)(NA + NT) * 20, dbtch, tmpseg, ND);
  final_logsoftmax_kernel<<<B, NM, 0, stream>>>(tmpseg, P(FIN), P(FIN + 1),
                                                ((float*)d_out) + B, NM);
}
